// MultiHeadAttention4D_60129542144792
// MI455X (gfx1250) — matrix operand order repaired and checked by reading, not compiled, not run on hardware
//
#include <hip/hip_runtime.h>
#include <hip/hip_bf16.h>

// MI455X / gfx1250 multi-head attention, WMMA f16 (f32 accumulate),
// async global->LDS staging of K/V tiles (ASYNCcnt path), double-buffered.
// Workspace layout (requires ~98MB):
//   [0,512K)      Wq16   [512K,1M) Wk16   [1M,1.5M) Wv16   [1.5M,2M) Wo16
//   [2M,34M)      Q16  (f16 [B,C,S,E])  -- reused in-place as Ctx16
//   [34M,66M)     K16  (f16 [B,C,S,E])
//   [66M,98M)     Vt16 (f16 [B,C,H,D,S])  per-head transposed V

typedef __attribute__((ext_vector_type(16))) _Float16 v16h;
typedef __attribute__((ext_vector_type(8)))  _Float16 v8h;
typedef __attribute__((ext_vector_type(8)))  float    v8f;

#define E_DIM  512
#define S_LEN  512
#define NH     8
#define HD     64
#define M_ROWS 32768   // B*C*S = 4*16*512

#define KPAD 72   // K-tile LDS row stride in halves (32 rows x 64 data halves)
#define VPAD 40   // V-tile LDS row stride in halves (64 rows x 32 data halves)

// ---- WMMA fragment loaders -------------------------------------------------
// A fragment, 16x32 f16 (MxK). Lane (r = lane&15, hi = lane>>4) holds row r:
// halves 0-7 = K[hi*8 .. +7], halves 8-15 = K[16+hi*8 .. +7].
__device__ __forceinline__ v16h load_a_f16(const _Float16* base, int ld) {
  const int lane = threadIdx.x & 31;
  const int r = lane & 15, hi = lane >> 4;
  const _Float16* p = base + (size_t)r * ld + hi * 8;
  v8h lo = *(const v8h*)p;
  v8h hh = *(const v8h*)(p + 16);
  v16h a;
#pragma unroll
  for (int i = 0; i < 8; ++i) { a[i] = lo[i]; a[i + 8] = hh[i]; }
  return a;
}

// A fragment from an f32 source (convert to f16 in registers).
__device__ __forceinline__ v16h load_a_f32(const float* base, int ld) {
  const int lane = threadIdx.x & 31;
  const int r = lane & 15, hi = lane >> 4;
  const float* p = base + (size_t)r * ld + hi * 8;
  v16h a;
#pragma unroll
  for (int i = 0; i < 8; ++i) {
    a[i]     = (_Float16)p[i];
    a[i + 8] = (_Float16)p[16 + i];
  }
  return a;
}

// B fragment, 32x16 f16 (KxN), given B-transposed rows: B[k][n] = rbase[n*ld+k].
// Lane (c, hi) holds column n=c: halves 0-7 = K[hi*8 .. +7], halves 8-15 = K[16+hi*8 .. +7].
__device__ __forceinline__ v16h load_b_f16(const _Float16* rbase, int ld) {
  const int lane = threadIdx.x & 31;
  const int c = lane & 15, hi = lane >> 4;
  const _Float16* p = rbase + (size_t)c * ld + hi * 8;
  v8h lo = *(const v8h*)p;
  v8h hh = *(const v8h*)(p + 16);
  v16h b;
#pragma unroll
  for (int i = 0; i < 8; ++i) { b[i] = lo[i]; b[i + 8] = hh[i]; }
  return b;
}

__device__ __forceinline__ v8f wmma32(v16h a, v16h b, v8f c) {
  return __builtin_amdgcn_wmma_f32_16x16x32_f16(
      /*neg_a=*/false, a, /*neg_b=*/false, b,
      /*c_mod=*/(short)0, c, /*reuse_a=*/false, /*reuse_b=*/false);
}

// ---- async global -> LDS (16B per lane), tracked with ASYNCcnt -------------
__device__ __forceinline__ void async_copy16(unsigned lds_off, const void* gptr) {
  __asm__ volatile("global_load_async_to_lds_b128 %0, %1, off"
                   :: "v"(lds_off), "v"(gptr) : "memory");
}
__device__ __forceinline__ unsigned lds_offset(const void* p) {
  // flat shared address = {SHARED_BASE, lds_byte_offset}; low 32 bits = offset
  return (unsigned)(size_t)p;
}

// ---- weight cast -----------------------------------------------------------
__global__ __launch_bounds__(256) void cast_f32_to_f16(const float* __restrict__ src,
                                                       _Float16* __restrict__ dst, int n) {
  int i = blockIdx.x * 256 + threadIdx.x;
  if (i < n) dst[i] = (_Float16)src[i];
}

// ---- projection GEMM: Y = A @ W^T + bias -----------------------------------
// MODE 0: A f32, Y f16 row-major [M,E]
// MODE 1: A f32, Y f16 transposed per head [B,C,H,D,S]
// MODE 2: A f16, Y f32 row-major [M,E]
// Wave computes a 16(M) x 128(N) tile; block = 8 waves stacked in M.
template <int MODE>
__global__ __launch_bounds__(256) void gemm_proj(const void* __restrict__ Av,
                                                 const _Float16* __restrict__ W16,
                                                 const float* __restrict__ bias,
                                                 void* __restrict__ Yv) {
  const int w    = threadIdx.x >> 5;
  const int lane = threadIdx.x & 31;
  const int c = lane & 15, hi = lane >> 4;
  const int m0 = blockIdx.x * 128 + w * 16;
  const int n0 = blockIdx.y * 128;

  v8f acc[8] = {};
  for (int kk = 0; kk < E_DIM / 32; ++kk) {
    v16h a;
    if (MODE == 2) a = load_a_f16((const _Float16*)Av + (size_t)m0 * E_DIM + kk * 32, E_DIM);
    else           a = load_a_f32((const float*)Av    + (size_t)m0 * E_DIM + kk * 32, E_DIM);
#pragma unroll
    for (int nt = 0; nt < 8; ++nt) {
      v16h b = load_b_f16(W16 + (size_t)(n0 + nt * 16) * E_DIM + kk * 32, E_DIM);
      acc[nt] = wmma32(a, b, acc[nt]);
    }
  }

#pragma unroll
  for (int nt = 0; nt < 8; ++nt) {
    const int n = n0 + nt * 16 + c;
    const float bn = bias[n];
#pragma unroll
    for (int r = 0; r < 8; ++r) {
      const int m = m0 + r + 8 * hi;
      const float val = acc[nt][r] + bn;
      if (MODE == 0) {
        ((_Float16*)Yv)[(size_t)m * E_DIM + n] = (_Float16)val;
      } else if (MODE == 1) {
        // dest[((bc*H + h)*D + d)*S + s] with n = h*64+d, bc = m>>9, s = m&511
        ((_Float16*)Yv)[((size_t)(m >> 9) * 512 + n) * 512 + (m & 511)] = (_Float16)val;
      } else {
        ((float*)Yv)[(size_t)m * E_DIM + n] = val;
      }
    }
  }
}

// ---- flash attention over one (bc, head, 128-row q-tile) per block ---------
// K/V tiles staged cooperatively into LDS via async loads, double-buffered.
__global__ __launch_bounds__(256) void flash_attn(const _Float16* __restrict__ Q16,
                                                  const _Float16* __restrict__ K16,
                                                  const _Float16* __restrict__ Vt16,
                                                  _Float16* __restrict__ Ctx16) {
  __shared__ __align__(16) _Float16 plds[8 * 16 * 32];       // P staging, 1KB/wave
  __shared__ __align__(16) _Float16 kbuf[2][32 * KPAD];      // 2 x 4.5KB
  __shared__ __align__(16) _Float16 vbuf[2][64 * VPAD];      // 2 x 5KB

  const int tid  = threadIdx.x;
  const int w    = tid >> 5;
  const int lane = tid & 31;
  const int c = lane & 15, hi = lane >> 4;
  const int bc = blockIdx.x >> 3;   // / NH
  const int h  = blockIdx.x & 7;    // % NH
  const int q0 = blockIdx.y * 128 + w * 16;

  const _Float16* Kh = K16  + (size_t)bc * S_LEN * E_DIM + h * HD;          // rows: s
  const _Float16* Vh = Vt16 + ((size_t)bc * NH + h) * (size_t)HD * S_LEN;   // rows: d

  // per-thread async staging assignment (issued by all 256 threads)
  const int krow = tid >> 3, koff = (tid & 7) * 16;   // 32 rows x 8 x 16B
  const int vrow = tid >> 2, voff = (tid & 3) * 16;   // 64 rows x 4 x 16B

  // ---- prologue: stage tile 0 into buffer 0
  async_copy16(lds_offset((const char*)&kbuf[0][0] + krow * (KPAD * 2) + koff),
               (const char*)(Kh + (size_t)krow * E_DIM) + koff);
  async_copy16(lds_offset((const char*)&vbuf[0][0] + vrow * (VPAD * 2) + voff),
               (const char*)(Vh + (size_t)vrow * S_LEN) + voff);

  const _Float16* Qb = Q16 + ((size_t)bc * S_LEN + q0) * E_DIM + h * HD;
  v16h aQ[2];
  aQ[0] = load_a_f16(Qb, E_DIM);
  aQ[1] = load_a_f16(Qb + 32, E_DIM);

  float mrun[8], lrun[8];
  v8f o[4] = {};
#pragma unroll
  for (int r = 0; r < 8; ++r) { mrun[r] = -3.0e38f; lrun[r] = 0.0f; }

  _Float16* myp = plds + w * (16 * 32);

  __asm__ volatile("s_wait_asynccnt 0x0" ::: "memory");
  __syncthreads();

  for (int t = 0; t < S_LEN / 32; ++t) {
    const _Float16* kt = kbuf[t & 1];
    const _Float16* vt = vbuf[t & 1];

    // prefetch tile t+1 into the other buffer (its readers passed last barrier)
    if (t + 1 < S_LEN / 32) {
      const int kv1 = (t + 1) * 32;
      async_copy16(lds_offset((const char*)&kbuf[(t + 1) & 1][0] + krow * (KPAD * 2) + koff),
                   (const char*)(Kh + (size_t)(kv1 + krow) * E_DIM) + koff);
      async_copy16(lds_offset((const char*)&vbuf[(t + 1) & 1][0] + vrow * (VPAD * 2) + voff),
                   (const char*)(Vh + (size_t)vrow * S_LEN + kv1) + voff);
    }

    // scores: S = Q @ K^T over D=64 (2 k-steps), 32 kv cols (2 n-tiles)
    v8f s0 = {}, s1 = {};
    s0 = wmma32(aQ[0], load_b_f16(kt, KPAD), s0);
    s0 = wmma32(aQ[1], load_b_f16(kt + 32, KPAD), s0);
    s1 = wmma32(aQ[0], load_b_f16(kt + 16 * KPAD, KPAD), s1);
    s1 = wmma32(aQ[1], load_b_f16(kt + 16 * KPAD + 32, KPAD), s1);

    // online softmax in f32; row stats reduced across the 16-lane half
#pragma unroll
    for (int r = 0; r < 8; ++r) {
      float x0 = s0[r] * 0.125f;   // 1/sqrt(64)
      float x1 = s1[r] * 0.125f;
      float vmax = fmaxf(x0, x1);
#pragma unroll
      for (int msk = 8; msk >= 1; msk >>= 1)
        vmax = fmaxf(vmax, __shfl_xor(vmax, msk, 32));
      const float mnew  = fmaxf(mrun[r], vmax);
      const float alpha = __expf(mrun[r] - mnew);
      const float p0 = __expf(x0 - mnew);
      const float p1 = __expf(x1 - mnew);
      float rs = p0 + p1;
#pragma unroll
      for (int msk = 8; msk >= 1; msk >>= 1)
        rs += __shfl_xor(rs, msk, 32);
      lrun[r] = lrun[r] * alpha + rs;
      mrun[r] = mnew;
      o[0][r] *= alpha; o[1][r] *= alpha; o[2][r] *= alpha; o[3][r] *= alpha;
      // stage P (C-layout) into this wave's LDS tile, row-major 16x32
      myp[(r + 8 * hi) * 32 + c]      = (_Float16)p0;
      myp[(r + 8 * hi) * 32 + c + 16] = (_Float16)p1;
    }
    __asm__ volatile("s_wait_dscnt 0x0" ::: "memory");  // intra-wave LDS RAW
    v16h aP = load_a_f16(myp, 32);   // re-read as A fragment (16x32, K=kv)

    // O += P @ V ; vt rows are d, K-contiguous in s -> direct B fragments
#pragma unroll
    for (int dt = 0; dt < 4; ++dt)
      o[dt] = wmma32(aP, load_b_f16(vt + dt * 16 * VPAD, VPAD), o[dt]);

    // next tile's LDS buffer must be filled and visible to all waves
    __asm__ volatile("s_wait_asynccnt 0x0" ::: "memory");
    __syncthreads();
  }

  // normalize and store ctx (f16, [B,C,S,E] head slice) -- aliases Q16 safely
  _Float16* Cb = Ctx16 + ((size_t)bc * S_LEN + q0) * E_DIM + h * HD;
#pragma unroll
  for (int r = 0; r < 8; ++r) {
    const float inv = 1.0f / lrun[r];
#pragma unroll
    for (int dt = 0; dt < 4; ++dt)
      Cb[(size_t)(r + 8 * hi) * E_DIM + dt * 16 + c] = (_Float16)(o[dt][r] * inv);
  }
}

// ---- host ------------------------------------------------------------------
extern "C" void kernel_launch(void* const* d_in, const int* in_sizes, int n_in,
                              void* d_out, int out_size, void* d_ws, size_t ws_size,
                              hipStream_t stream) {
  const float* q  = (const float*)d_in[0];
  const float* k  = (const float*)d_in[1];
  const float* v  = (const float*)d_in[2];
  const float* Wq = (const float*)d_in[3];
  const float* bq = (const float*)d_in[4];
  const float* Wk = (const float*)d_in[5];
  const float* bk = (const float*)d_in[6];
  const float* Wv = (const float*)d_in[7];
  const float* bv = (const float*)d_in[8];
  const float* Wo = (const float*)d_in[9];
  const float* bo = (const float*)d_in[10];
  float* out = (float*)d_out;

  char* ws = (char*)d_ws;
  const size_t MB = 1024 * 1024;
  _Float16* Wq16 = (_Float16*)(ws);
  _Float16* Wk16 = (_Float16*)(ws + 512 * 1024);
  _Float16* Wv16 = (_Float16*)(ws + 1 * MB);
  _Float16* Wo16 = (_Float16*)(ws + MB + 512 * 1024);
  _Float16* Q16  = (_Float16*)(ws + 2 * MB);    // reused as Ctx16
  _Float16* K16  = (_Float16*)(ws + 34 * MB);
  _Float16* Vt16 = (_Float16*)(ws + 66 * MB);

  const int nw = E_DIM * E_DIM;
  dim3 gc((nw + 255) / 256);
  cast_f32_to_f16<<<gc, 256, 0, stream>>>(Wq, Wq16, nw);
  cast_f32_to_f16<<<gc, 256, 0, stream>>>(Wk, Wk16, nw);
  cast_f32_to_f16<<<gc, 256, 0, stream>>>(Wv, Wv16, nw);
  cast_f32_to_f16<<<gc, 256, 0, stream>>>(Wo, Wo16, nw);

  dim3 gp(M_ROWS / 128, E_DIM / 128);   // 256 x 4 blocks, 256 threads
  gemm_proj<0><<<gp, 256, 0, stream>>>(q, Wq16, bq, Q16);
  gemm_proj<0><<<gp, 256, 0, stream>>>(k, Wk16, bk, K16);
  gemm_proj<1><<<gp, 256, 0, stream>>>(v, Wv16, bv, Vt16);

  flash_attn<<<dim3(4 * 16 * NH, S_LEN / 128), 256, 0, stream>>>(Q16, K16, Vt16, Q16);

  gemm_proj<2><<<gp, 256, 0, stream>>>(Q16, Wo16, bo, out);
}
